// StrokeVQVAE_7533372637766
// MI455X (gfx1250) — compile-verified, matrix operand order checked by reading, not run on hardware
//
#include <hip/hip_runtime.h>

// ---------- feature detection ----------
#if defined(__has_builtin)
# if __has_builtin(__builtin_amdgcn_global_load_async_to_lds_b128)
#  define HAVE_ASYNC_LDS 1
# else
#  define HAVE_ASYNC_LDS 0
# endif
# if __has_builtin(__builtin_amdgcn_s_wait_asynccnt)
#  define HAVE_WAIT_ASYNC 1
# else
#  define HAVE_WAIT_ASYNC 0
# endif
#else
# define HAVE_ASYNC_LDS 0
# define HAVE_WAIT_ASYNC 0
#endif

// ---------- types ----------
typedef __attribute__((ext_vector_type(16))) __bf16 v16bf;
typedef __attribute__((ext_vector_type(8)))  float  v8f;
typedef __attribute__((ext_vector_type(4)))  int    v4i;

typedef __attribute__((address_space(1))) v4i gv4i;   // global
typedef __attribute__((address_space(3))) v4i lv4i;   // LDS

union FragBF {
    v16bf v;
    uint4 q[2];
};

__device__ __forceinline__ unsigned short f2bf(float f) {
    unsigned int u = __float_as_uint(f);
    u += 0x7FFFu + ((u >> 16) & 1u);   // round-to-nearest-even
    return (unsigned short)(u >> 16);
}
__device__ __forceinline__ float bf2f(unsigned short h) {
    return __uint_as_float(((unsigned int)h) << 16);
}

// global -> LDS copy of 16 bytes (async path avoids the VGPR bounce)
__device__ __forceinline__ void copy_g2l_b128(const unsigned short* g, unsigned short* l) {
#if HAVE_ASYNC_LDS
    __builtin_amdgcn_global_load_async_to_lds_b128((gv4i*)g, (lv4i*)l, 0, 0);
#else
    *reinterpret_cast<uint4*>(l) = *reinterpret_cast<const uint4*>(g);
#endif
}
__device__ __forceinline__ void wait_async_copies() {
#if HAVE_ASYNC_LDS
# if HAVE_WAIT_ASYNC
    __builtin_amdgcn_s_wait_asynccnt(0);
# else
    asm volatile("s_wait_asynccnt 0x0" ::: "memory");
# endif
#endif
}

// ---------- constants ----------
#define BATCH    32768
#define IN_DIM   128
#define HID      512
#define TOK_DIM  512
#define NTOK     131072          // B*T
#define DEMB     128
#define KCODE    1024

// ---------- small utility kernels ----------
__global__ void zero_k(float* counts, float* accum) {
    int i = blockIdx.x * blockDim.x + threadIdx.x;
    if (i < KCODE) counts[i] = 0.f;
    if (i < 8)     accum[i]  = 0.f;
}

__global__ void f32_to_bf16_k(const float* __restrict__ in,
                              unsigned short* __restrict__ out, int n) {
    int i = blockIdx.x * blockDim.x + threadIdx.x;
    if (i < n) out[i] = f2bf(in[i]);
}

// Pack a K x N (row-major) fp32 weight matrix into bf16 B-fragment tiles.
// Tile (nt, kt) at flat index nt*kTiles + kt; within a tile:
//   addr = tile*512 + chunk*256 + lane*8 + j
//   lane 0..15  : col = nt*16 + lane,      k = kt*32 + chunk*16 + j
//   lane 16..31 : col = nt*16 + (lane-16), k = kt*32 + chunk*16 + 8 + j
// transpose=1: source is (N rows x K cols) row-major (embedding^T).
__global__ void pack_w_k(const float* __restrict__ W,
                         unsigned short* __restrict__ out,
                         int K, int N, int transpose) {
    int e = blockIdx.x * blockDim.x + threadIdx.x;
    int total = K * N;
    if (e >= total) return;
    int tile   = e >> 9;
    int within = e & 511;
    int chunk  = within >> 8;
    int lane   = (within >> 3) & 31;
    int j      = within & 7;
    int kTiles = K >> 5;
    int nt = tile / kTiles;
    int kt = tile - nt * kTiles;
    int n = nt * 16 + (lane & 15);
    int k = kt * 32 + chunk * 16 + (lane >> 4) * 8 + j;
    float v = transpose ? W[(size_t)n * K + k] : W[(size_t)k * N + n];
    out[e] = f2bf(v);
}

__global__ void enorm_k(const float* __restrict__ emb, float* __restrict__ enorm2) {
    int c = blockIdx.x * blockDim.x + threadIdx.x;
    if (c < KCODE) {
        const float* p = emb + (size_t)c * DEMB;
        float s = 0.f;
        for (int d = 0; d < DEMB; ++d) s += p[d] * p[d];
        enorm2[c] = s;
    }
}

// ---------- fused GEMM + bias + (SiLU), LDS-staged B panel ----------
// Block = 8 waves = 128 M-rows x 64 N-cols. The 64-col B panel (K*64 bf16,
// contiguous in the packed layout) is staged once into LDS via async copies
// and shared by all 8 waves. Each wave: 16x64 strip, 4 WMMAs per K-step.
__global__ __launch_bounds__(256) void gemm_bias_act_k(
    const unsigned short* __restrict__ A,
    const unsigned short* __restrict__ Wp,
    const float* __restrict__ bias,
    unsigned short* __restrict__ outB,
    float* __restrict__ outF,
    int M, int N, int K, int doSilu)
{
    extern __shared__ unsigned short sB[];   // K*64 halves
    const int tid  = threadIdx.x;
    const int lane = tid & 31;
    const int wave = tid >> 5;
    const int nStrips = N >> 6;
    const int nStrip = blockIdx.x % nStrips;
    const int mBlock = blockIdx.x / nStrips;
    const int kTiles = K >> 5;
    const int half = lane >> 4;
    const int l15  = lane & 15;

    // ---- stage B panel: contiguous region of 4*kTiles tiles ----
    const unsigned short* gsrc = Wp + (size_t)nStrip * 4 * kTiles * 512;
    const int panelHalves = K * 64;
    for (int h = tid * 8; h < panelHalves; h += 256 * 8)
        copy_g2l_b128(gsrc + h, sB + h);
    wait_async_copies();
    __syncthreads();

    // ---- compute ----
    const int m0 = (mBlock * 8 + wave) * 16;
    v8f acc[4] = {};
    const unsigned short* aBase = A + (size_t)(m0 + l15) * K + half * 8;
    for (int kt = 0; kt < kTiles; ++kt) {
        FragBF af;
        const unsigned short* ap = aBase + kt * 32;
        af.q[0] = *reinterpret_cast<const uint4*>(ap);
        af.q[1] = *reinterpret_cast<const uint4*>(ap + 16);
#pragma unroll
        for (int j = 0; j < 4; ++j) {
            const int tb = ((j * kTiles + kt) << 9) + lane * 8;
            FragBF bf;
            bf.q[0] = *reinterpret_cast<const uint4*>(sB + tb);
            bf.q[1] = *reinterpret_cast<const uint4*>(sB + tb + 256);
            acc[j] = __builtin_amdgcn_wmma_f32_16x16x32_bf16(
                false, af.v, false, bf.v, (short)0, acc[j], false, false);
        }
    }

#pragma unroll
    for (int j = 0; j < 4; ++j) {
        const int n = nStrip * 64 + j * 16 + l15;
        const float bi = bias[n];
        v8f a = acc[j];
#pragma unroll
        for (int r = 0; r < 8; ++r) {
            const int m = m0 + half * 8 + r;
            float v = a[r] + bi;
            if (doSilu) v = v / (1.f + __expf(-v));
            const size_t o = (size_t)m * N + n;
            if (outB) outB[o] = f2bf(v);
            if (outF) outF[o] = v;
        }
    }
}

// ---------- VQ argmin, LDS-staged codebook chunks ----------
// Block = 8 waves x 16 tokens. All waves scan the same 1024 codes, so the
// packed codebook is streamed through LDS in 64-code (16 KB) chunks shared
// by the whole block. score(c) = |e_c|^2 - 2 z.e_c (|z|^2 dropped).
__global__ __launch_bounds__(256) void vq_argmin_k(
    const unsigned short* __restrict__ Z,      // (NTOK x 128) bf16
    const unsigned short* __restrict__ Ep,     // packed embedding^T (K=128,N=1024)
    const float* __restrict__ enorm2,
    int* __restrict__ idxOut,
    float* __restrict__ idxF)
{
    __shared__ unsigned short sE[64 * DEMB];   // 8192 halves = 16 KB
    const int tid  = threadIdx.x;
    const int lane = tid & 31;
    const int wave = tid >> 5;
    const int m0 = (blockIdx.x * 8 + wave) * 16;
    const int half = lane >> 4;
    const int l15  = lane & 15;

    // Hoist the 4 A fragments (16 tokens x 128 dims) -- reused for all codes.
    FragBF af[4];
    const unsigned short* aBase = Z + (size_t)(m0 + l15) * DEMB + half * 8;
#pragma unroll
    for (int kt = 0; kt < 4; ++kt) {
        const unsigned short* ap = aBase + kt * 32;
        af[kt].q[0] = *reinterpret_cast<const uint4*>(ap);
        af[kt].q[1] = *reinterpret_cast<const uint4*>(ap + 16);
    }

    float bestV[8];
    int   bestI[8];
#pragma unroll
    for (int r = 0; r < 8; ++r) { bestV[r] = 3.4e38f; bestI[r] = 0; }

    for (int chunk = 0; chunk < 16; ++chunk) {
        __syncthreads();                        // previous chunk fully consumed
        const unsigned short* gsrc = Ep + ((size_t)chunk << 13);  // 8192 halves/chunk
        for (int h = tid * 8; h < 8192; h += 256 * 8)
            copy_g2l_b128(gsrc + h, sE + h);
        wait_async_copies();
        __syncthreads();

#pragma unroll
        for (int j = 0; j < 4; ++j) {
            v8f acc = {};
#pragma unroll
            for (int kt = 0; kt < 4; ++kt) {
                const int tb = (((j << 2) + kt) << 9) + lane * 8;
                FragBF bf;
                bf.q[0] = *reinterpret_cast<const uint4*>(sE + tb);
                bf.q[1] = *reinterpret_cast<const uint4*>(sE + tb + 256);
                acc = __builtin_amdgcn_wmma_f32_16x16x32_bf16(
                    false, af[kt].v, false, bf.v, (short)0, acc, false, false);
            }
            const int c = chunk * 64 + j * 16 + l15;
            const float en = enorm2[c];
#pragma unroll
            for (int r = 0; r < 8; ++r) {
                const float s = en - 2.f * acc[r];
                if (s < bestV[r]) { bestV[r] = s; bestI[r] = c; }
            }
        }
    }

    // reduce across the 16 lanes sharing a token (xor 8,4,2,1 stays in-half)
#pragma unroll
    for (int r = 0; r < 8; ++r) {
        float v = bestV[r];
        int   bi = bestI[r];
#pragma unroll
        for (int s = 8; s >= 1; s >>= 1) {
            float ov = __shfl_xor(v, s, 32);
            int   oi = __shfl_xor(bi, s, 32);
            if (ov < v || (ov == v && oi < bi)) { v = ov; bi = oi; }
        }
        if (l15 == 0) {
            const int t = m0 + half * 8 + r;
            idxOut[t] = bi;
            idxF[t] = (float)bi;
        }
    }
}

__global__ void hist_k(const int* __restrict__ idx, float* __restrict__ counts) {
    int i = blockIdx.x * blockDim.x + threadIdx.x;
    if (i < NTOK) atomicAdd(counts + idx[i], 1.f);   // integer-valued: exact
}

// Gather quantized vectors (in place over z), accumulate sum((q - z)^2)
// into per-block partials (deterministic second-pass reduce).
__global__ __launch_bounds__(256) void quantize_k(
    const int* __restrict__ idx,
    const float* __restrict__ emb,
    unsigned short* __restrict__ ZQ,     // in: z bf16, out: quantized bf16
    float* __restrict__ partials)
{
    float local = 0.f;
    const size_t total = (size_t)NTOK * DEMB;
    for (size_t e = (size_t)blockIdx.x * blockDim.x + threadIdx.x; e < total;
         e += (size_t)gridDim.x * blockDim.x) {
        const int t = (int)(e >> 7);
        const int d = (int)(e & 127);
        const float q = emb[(size_t)idx[t] * DEMB + d];
        const float z = bf2f(ZQ[e]);
        const float df = q - z;
        local += df * df;
        ZQ[e] = f2bf(q);
    }
    __shared__ float sm[256];
    sm[threadIdx.x] = local;
    __syncthreads();
    for (int s = 128; s > 0; s >>= 1) {
        if (threadIdx.x < s) sm[threadIdx.x] += sm[threadIdx.x + s];
        __syncthreads();
    }
    if (threadIdx.x == 0) partials[blockIdx.x] = sm[0];
}

__global__ __launch_bounds__(256) void recon_loss_k(
    const float* __restrict__ recon,
    const float* __restrict__ strokes,
    float* __restrict__ partials)
{
    float local = 0.f;
    const size_t total = (size_t)BATCH * IN_DIM;
    for (size_t e = (size_t)blockIdx.x * blockDim.x + threadIdx.x; e < total;
         e += (size_t)gridDim.x * blockDim.x) {
        const float df = recon[e] - strokes[e];
        local += df * df;
    }
    __shared__ float sm[256];
    sm[threadIdx.x] = local;
    __syncthreads();
    for (int s = 128; s > 0; s >>= 1) {
        if (threadIdx.x < s) sm[threadIdx.x] += sm[threadIdx.x + s];
        __syncthreads();
    }
    if (threadIdx.x == 0) partials[blockIdx.x] = sm[0];
}

__global__ __launch_bounds__(1024) void reduce_k(const float* __restrict__ in,
                                                 int n, float* __restrict__ outv) {
    __shared__ float sm[1024];
    float s = 0.f;
    for (int i = threadIdx.x; i < n; i += 1024) s += in[i];
    sm[threadIdx.x] = s;
    __syncthreads();
    for (int st = 512; st > 0; st >>= 1) {
        if (threadIdx.x < st) sm[threadIdx.x] += sm[threadIdx.x + st];
        __syncthreads();
    }
    if (threadIdx.x == 0) *outv = sm[0];
}

__global__ __launch_bounds__(1024) void finalize_k(const float* __restrict__ counts,
                                                   const float* __restrict__ accum,
                                                   float* __restrict__ out) {
    __shared__ float sm[1024];
    const int c = threadIdx.x;
    const float p = counts[c] * (1.f / (float)NTOK);
    sm[c] = p * logf(p + 1e-10f);
    __syncthreads();
    for (int s = 512; s > 0; s >>= 1) {
        if (c < s) sm[c] += sm[c + s];
        __syncthreads();
    }
    if (c == 0) {
        const float perp = expf(-sm[0]);
        const float vq_mean = accum[0] * (1.f / ((float)NTOK * (float)DEMB));
        const float vq_loss = 1.25f * vq_mean;               // commit 0.25 + codebook 1.0
        const float recon_loss = accum[1] * (1.f / ((float)BATCH * (float)IN_DIM));
        out[0] = recon_loss + vq_loss;
        out[1] = recon_loss;
        out[2] = vq_loss;
        out[3] = perp;
    }
}

// ---------- host ----------
extern "C" void kernel_launch(void* const* d_in, const int* in_sizes, int n_in,
                              void* d_out, int out_size, void* d_ws, size_t ws_size,
                              hipStream_t stream) {
    const float* strokes = (const float*)d_in[0];
    const float* We1 = (const float*)d_in[1];  const float* be1 = (const float*)d_in[2];
    const float* We2 = (const float*)d_in[3];  const float* be2 = (const float*)d_in[4];
    const float* We3 = (const float*)d_in[5];  const float* be3 = (const float*)d_in[6];
    const float* Wd1 = (const float*)d_in[7];  const float* bd1 = (const float*)d_in[8];
    const float* Wd2 = (const float*)d_in[9];  const float* bd2 = (const float*)d_in[10];
    const float* Wd3 = (const float*)d_in[11]; const float* bd3 = (const float*)d_in[12];
    const float* emb = (const float*)d_in[13];

    float* out = (float*)d_out;
    float* recon   = out + 4;
    float* idxFout = out + 4 + (size_t)BATCH * IN_DIM;

    // workspace carve-up (all offsets 256B aligned)
    char* ws = (char*)d_ws;
    size_t off = 0;
    auto carve = [&](size_t bytes) -> void* {
        void* p = ws + off;
        off = (off + bytes + 255) & ~(size_t)255;
        return p;
    };
    float* accum  = (float*)carve(8 * sizeof(float));
    float* counts = (float*)carve(KCODE * sizeof(float));
    float* enorm2 = (float*)carve(KCODE * sizeof(float));
    int*   idx_i  = (int*)carve(NTOK * sizeof(int));
    float* partials = (float*)carve(8192 * sizeof(float));
    unsigned short* we1p = (unsigned short*)carve((size_t)IN_DIM * HID * 2);
    unsigned short* we2p = (unsigned short*)carve((size_t)HID * HID * 2);
    unsigned short* we3p = (unsigned short*)carve((size_t)HID * TOK_DIM * 2);
    unsigned short* wd1p = (unsigned short*)carve((size_t)TOK_DIM * HID * 2);
    unsigned short* wd2p = (unsigned short*)carve((size_t)HID * HID * 2);
    unsigned short* wd3p = (unsigned short*)carve((size_t)HID * IN_DIM * 2);
    unsigned short* ep   = (unsigned short*)carve((size_t)DEMB * KCODE * 2);
    unsigned short* sbf  = (unsigned short*)carve((size_t)BATCH * IN_DIM * 2);
    unsigned short* actA = (unsigned short*)carve((size_t)BATCH * HID * 2);
    unsigned short* actB = (unsigned short*)carve((size_t)BATCH * HID * 2);
    unsigned short* zq   = (unsigned short*)carve((size_t)BATCH * TOK_DIM * 2);
    (void)ws_size; (void)n_in; (void)in_sizes; (void)out_size;

    // init
    zero_k<<<4, 256, 0, stream>>>(counts, accum);
    f32_to_bf16_k<<<(BATCH * IN_DIM) / 256, 256, 0, stream>>>(strokes, sbf, BATCH * IN_DIM);

    // pack weights + embedding^T into WMMA B-fragment layout
    pack_w_k<<<(IN_DIM * HID) / 256, 256, 0, stream>>>(We1, we1p, IN_DIM, HID, 0);
    pack_w_k<<<(HID * HID) / 256, 256, 0, stream>>>(We2, we2p, HID, HID, 0);
    pack_w_k<<<(HID * TOK_DIM) / 256, 256, 0, stream>>>(We3, we3p, HID, TOK_DIM, 0);
    pack_w_k<<<(TOK_DIM * HID) / 256, 256, 0, stream>>>(Wd1, wd1p, TOK_DIM, HID, 0);
    pack_w_k<<<(HID * HID) / 256, 256, 0, stream>>>(Wd2, wd2p, HID, HID, 0);
    pack_w_k<<<(HID * IN_DIM) / 256, 256, 0, stream>>>(Wd3, wd3p, HID, IN_DIM, 0);
    pack_w_k<<<(DEMB * KCODE) / 256, 256, 0, stream>>>(emb, ep, DEMB, KCODE, 1);
    enorm_k<<<4, 256, 0, stream>>>(emb, enorm2);

    // encoder  (dynamic LDS = K*64 bf16 bytes)
    gemm_bias_act_k<<<2048, 256, (size_t)IN_DIM * 64 * 2, stream>>>(
        sbf,  we1p, be1, actA, nullptr, BATCH, HID, IN_DIM, 1);
    gemm_bias_act_k<<<2048, 256, (size_t)HID * 64 * 2, stream>>>(
        actA, we2p, be2, actB, nullptr, BATCH, HID, HID, 1);
    gemm_bias_act_k<<<2048, 256, (size_t)HID * 64 * 2, stream>>>(
        actB, we3p, be3, zq,   nullptr, BATCH, TOK_DIM, HID, 0);

    // VQ
    vq_argmin_k<<<1024, 256, 0, stream>>>(zq, ep, enorm2, idx_i, idxFout);
    hist_k<<<NTOK / 256, 256, 0, stream>>>(idx_i, counts);
    quantize_k<<<8192, 256, 0, stream>>>(idx_i, emb, zq, partials);
    reduce_k<<<1, 1024, 0, stream>>>(partials, 8192, accum + 0);

    // decoder (zq now holds quantized bf16 == q_st forward value)
    gemm_bias_act_k<<<2048, 256, (size_t)TOK_DIM * 64 * 2, stream>>>(
        zq,   wd1p, bd1, actA, nullptr, BATCH, HID, TOK_DIM, 1);
    gemm_bias_act_k<<<2048, 256, (size_t)HID * 64 * 2, stream>>>(
        actA, wd2p, bd2, actB, nullptr, BATCH, HID, HID, 1);
    gemm_bias_act_k<<<512,  256, (size_t)HID * 64 * 2, stream>>>(
        actB, wd3p, bd3, nullptr, recon, BATCH, IN_DIM, HID, 0);

    // losses + scalars
    recon_loss_k<<<2048, 256, 0, stream>>>(recon, strokes, partials);
    reduce_k<<<1, 1024, 0, stream>>>(partials, 2048, accum + 1);
    finalize_k<<<1, 1024, 0, stream>>>(counts, accum, out);
}
